// PerTokenSparseMoE_5970004542167
// MI455X (gfx1250) — matrix operand order repaired and checked by reading, not compiled, not run on hardware
//
#include <hip/hip_runtime.h>

// MoE forward for gfx1250 (MI455X): bf16 WMMA (f32 accum), LDS-staged tiles,
// bank-conflict-free padded LDS layouts, global_prefetch_b8 chunk prefetch.
//   y[b,t,:] = sum_e relu(x.gw+gb)[b,t,e] * ( gelu(x W1[t,e] + b1) W2[t,e] + b2 )
// Block = 256 threads (8 wave32) handles (t, 64-token tile), loops e and H-chunks.

#define TT 16
#define EE 4
#define DD 256
#define HH 1024
#define BB 1024
#define BM 64       // tokens per block
#define HC 32       // H-chunk width
#define XS (DD + 8) // padded k-stride (ushorts) for 256-deep rows: 528B -> bank stride 4
#define HS (HC + 8) // padded k-stride (ushorts) for 32-deep rows:  80B -> bank stride 20

typedef __bf16 bf16_t;
typedef bf16_t v16bf __attribute__((ext_vector_type(16)));
typedef float  v8f   __attribute__((ext_vector_type(8)));

__device__ __forceinline__ unsigned short f2bfu(float f) {
  unsigned u = __builtin_bit_cast(unsigned, f);
  u += 0x7FFFu + ((u >> 16) & 1u);          // round-to-nearest-even
  return (unsigned short)(u >> 16);
}

__device__ __forceinline__ float gelu_tanh(float x) {
  float x3 = x * x * x;
  return 0.5f * x * (1.0f + tanhf(0.7978845608028654f * (x + 0.044715f * x3)));
}

// A fragment (16x32 bf16): lanes 0-15 -> M=0..15 (K base 0), lanes 16-31 same M, K base +8.
__device__ __forceinline__ v16bf load_a(const unsigned short* p, int m0, int k0,
                                        int stride, int lane) {
  const unsigned short* row = p + (m0 + (lane & 15)) * stride;
  const int kh = (lane >> 4) << 3;          // 0 or 8
  v16bf a;
  unsigned* au = (unsigned*)&a;
#pragma unroll
  for (int j = 0; j < 4; ++j) au[j]     = *(const unsigned*)(row + k0 + kh + 2 * j);
#pragma unroll
  for (int j = 0; j < 4; ++j) au[4 + j] = *(const unsigned*)(row + k0 + 16 + kh + 2 * j);
  return a;
}

// B fragment (32x16 bf16): lanes 0-15 -> N=0..15 (K base 0), lanes 16-31 same N, K base +8.
// LDS buffer stored [col][k] so K-pairs are contiguous u32/b128 loads.
__device__ __forceinline__ v16bf load_b(const unsigned short* p, int n0, int k0,
                                        int stride, int lane) {
  const unsigned short* col = p + (n0 + (lane & 15)) * stride;
  const int kh = (lane >> 4) << 3;          // 0 or 8
  v16bf b;
  unsigned* bu = (unsigned*)&b;
#pragma unroll
  for (int j = 0; j < 4; ++j) bu[j]     = *(const unsigned*)(col + k0 + kh + 2 * j);
#pragma unroll
  for (int j = 0; j < 4; ++j) bu[4 + j] = *(const unsigned*)(col + k0 + 16 + kh + 2 * j);
  return b;
}

__global__ __launch_bounds__(256)
void moe_wmma_kernel(const float* __restrict__ X,  const float* __restrict__ W1,
                     const float* __restrict__ B1, const float* __restrict__ W2,
                     const float* __restrict__ B2, const float* __restrict__ GW,
                     const float* __restrict__ GB, float* __restrict__ Y) {
  __shared__ __align__(16) unsigned short xbf[BM * XS];   // 33.0 KB  x tile, [row][k]
  __shared__ __align__(16) unsigned short wcb[DD * HS];   // 20.0 KB  weight chunk, [col][k]
  __shared__ __align__(16) unsigned short hbf[BM * HS];   // 5.0 KB   gated gelu(h), [row][k]
  __shared__ float gateS[BM * EE];                        // 1.0 KB

  const int t    = blockIdx.y;
  const int row0 = blockIdx.x * BM;
  const int tid  = threadIdx.x;
  const int lane = tid & 31;
  const int wave = tid >> 5;
  const int wm   = wave & 3;                // M tile (16 rows)
  const int wn   = wave >> 2;               // 0..1
  const int hi8  = (lane >> 4) << 3;        // C-tile row offset for upper lane half
  const int l15  = lane & 15;

  // ---- stage x tile as bf16 (reused 128x) ----
#pragma unroll
  for (int i = 0; i < 16; ++i) {
    int idx4 = tid + (i << 8);              // 0..4095 float4s
    int r  = idx4 >> 6;
    int c4 = (idx4 & 63) << 2;
    const float4 v = *(const float4*)(X + ((size_t)(row0 + r) * TT + t) * DD + c4);
    unsigned p0 = (unsigned)f2bfu(v.x) | ((unsigned)f2bfu(v.y) << 16);
    unsigned p1 = (unsigned)f2bfu(v.z) | ((unsigned)f2bfu(v.w) << 16);
    *(unsigned*)(&xbf[r * XS + c4])     = p0;
    *(unsigned*)(&xbf[r * XS + c4 + 2]) = p1;
  }
  __syncthreads();

  // ---- router gates: one (token, expert) per thread ----
  {
    int r = tid >> 2, e = tid & 3;
    float acc = GB[t * EE + e];
    const float* xr = X + ((size_t)(row0 + r) * TT + t) * DD;
    const float* gc = GW + (size_t)t * DD * EE + e;
    for (int d = 0; d < DD; ++d) acc = fmaf(xr[d], gc[(size_t)d * EE], acc);
    gateS[r * EE + e] = fmaxf(acc, 0.0f);
  }
  __syncthreads();

  const v8f vzero = {0.f, 0.f, 0.f, 0.f, 0.f, 0.f, 0.f, 0.f};
  v8f accY[8];                              // this wave's 16x128 slice of y
#pragma unroll
  for (int j = 0; j < 8; ++j) accY[j] = vzero;

  for (int e = 0; e < EE; ++e) {
    const float* w1e = W1 + (size_t)(t * EE + e) * DD * HH;
    const float* w2e = W2 + (size_t)(t * EE + e) * HH * DD;
    float g[8];
#pragma unroll
    for (int r = 0; r < 8; ++r) g[r] = gateS[(wm * 16 + r + hi8) * EE + e];

    for (int hc = 0; hc < HH / HC; ++hc) {
      const int hbase = hc * HC;

      // stage W1[:, hbase:hbase+HC] -> wcb[col][k=0..255] (bf16, transposed)
#pragma unroll
      for (int i = 0; i < 8; ++i) {
        int idx4 = tid + (i << 8);          // 0..2047
        int k  = idx4 >> 3;                 // 0..255
        int c4 = (idx4 & 7) << 2;           // 0..31
        const float4 v = *(const float4*)(w1e + (size_t)k * HH + hbase + c4);
        wcb[(c4 + 0) * XS + k] = f2bfu(v.x);
        wcb[(c4 + 1) * XS + k] = f2bfu(v.y);
        wcb[(c4 + 2) * XS + k] = f2bfu(v.z);
        wcb[(c4 + 3) * XS + k] = f2bfu(v.w);
      }
      // prefetch this chunk's W2 slab (32 rows x 256 cols = 256 x 128B lines)
      // while GEMM1 runs: one line per thread.
      __builtin_prefetch(w2e + (size_t)(hbase + (tid >> 3)) * DD + ((tid & 7) << 5), 0, 3);
      __syncthreads();

      // GEMM1: h 16x16 tile (wm, wn) of the 64xHC chunk; K = 256
      v8f accH = vzero;
#pragma unroll
      for (int k0 = 0; k0 < DD; k0 += 32) {
        v16bf a = load_a(xbf, wm * 16, k0, XS, lane);
        v16bf b = load_b(wcb, wn * 16, k0, XS, lane);
        accH = __builtin_amdgcn_wmma_f32_16x16x32_bf16(false, a, false, b,
                                                       (short)0, accH, false, false);
      }

      // bias + gelu + fold router gate, write bf16 h chunk
      {
        const int   col  = hbase + wn * 16 + l15;
        const float bias = B1[(size_t)(t * EE + e) * HH + col];
#pragma unroll
        for (int r = 0; r < 8; ++r) {
          float v = gelu_tanh(accH[r] + bias) * g[r];
          hbf[(wm * 16 + r + hi8) * HS + wn * 16 + l15] = f2bfu(v);
        }
      }
      __syncthreads();                      // wcb reads + hbf writes complete

      // stage W2[hbase:hbase+HC, :] -> wcb[col][k=0..31] (bf16, transposed)
#pragma unroll
      for (int i = 0; i < 8; ++i) {
        int idx4 = tid + (i << 8);          // 0..2047
        int k  = idx4 >> 6;                 // 0..31
        int c4 = (idx4 & 63) << 2;          // 0..255
        const float4 v = *(const float4*)(w2e + (size_t)(hbase + k) * DD + c4);
        wcb[(c4 + 0) * HS + k] = f2bfu(v.x);
        wcb[(c4 + 1) * HS + k] = f2bfu(v.y);
        wcb[(c4 + 2) * HS + k] = f2bfu(v.z);
        wcb[(c4 + 3) * HS + k] = f2bfu(v.w);
      }
      // prefetch next chunk's W1 slab (256 rows x 32 cols = 256 x 128B lines)
      // while GEMM2 runs: one line per thread.
      if (hc + 1 < HH / HC)
        __builtin_prefetch(w1e + (size_t)tid * HH + hbase + HC, 0, 3);
      __syncthreads();

      // GEMM2 partial: y(16x128 per wave) += h_chunk(16x32) * W2_chunk(32x128)
      v16bf a2 = load_a(hbf, wm * 16, 0, HS, lane);
#pragma unroll
      for (int j = 0; j < 8; ++j) {
        v16bf b2f = load_b(wcb, wn * 128 + j * 16, 0, HS, lane);
        accY[j] = __builtin_amdgcn_wmma_f32_16x16x32_bf16(false, a2, false, b2f,
                                                          (short)0, accY[j], false, false);
      }
      __syncthreads();                      // before next chunk overwrites wcb/hbf
    }
  }

  // ---- epilogue: + sum_e gate_e * b2_e, store y ----
  float gAll[8][EE];
#pragma unroll
  for (int r = 0; r < 8; ++r)
#pragma unroll
    for (int e = 0; e < EE; ++e) gAll[r][e] = gateS[(wm * 16 + r + hi8) * EE + e];

#pragma unroll
  for (int j = 0; j < 8; ++j) {
    const int col = wn * 128 + j * 16 + l15;
    float bv[EE];
#pragma unroll
    for (int e = 0; e < EE; ++e) bv[e] = B2[(size_t)(t * EE + e) * DD + col];
#pragma unroll
    for (int r = 0; r < 8; ++r) {
      float v = accY[j][r];
#pragma unroll
      for (int e = 0; e < EE; ++e) v = fmaf(gAll[r][e], bv[e], v);
      Y[((size_t)(row0 + wm * 16 + r + hi8) * TT + t) * DD + col] = v;
    }
  }
}

extern "C" void kernel_launch(void* const* d_in, const int* in_sizes, int n_in,
                              void* d_out, int out_size, void* d_ws, size_t ws_size,
                              hipStream_t stream) {
  (void)in_sizes; (void)n_in; (void)out_size; (void)d_ws; (void)ws_size;
  const float* X  = (const float*)d_in[0];
  const float* W1 = (const float*)d_in[1];
  const float* B1 = (const float*)d_in[2];
  const float* W2 = (const float*)d_in[3];
  const float* B2 = (const float*)d_in[4];
  const float* GW = (const float*)d_in[5];
  const float* GB = (const float*)d_in[6];
  float* Y = (float*)d_out;

  dim3 grid(BB / BM, TT);                   // (16, 16) blocks, 256 threads each
  moe_wmma_kernel<<<grid, 256, 0, stream>>>(X, W1, B1, W2, B2, GW, GB, Y);
}
